// MemoryEfficientGNNRegression_31636729102835
// MI455X (gfx1250) — matrix operand order repaired and checked by reading, not compiled, not run on hardware
//
#include <hip/hip_runtime.h>

#define N_STAY 200000
#define N_PAT  100000
#define NEDGE  1000000
#define H 64

typedef __attribute__((ext_vector_type(16))) __bf16 v16bf;
typedef __attribute__((ext_vector_type(8)))  float  v8f;

#define WMMA_BF16(A, B, C) \
  __builtin_amdgcn_wmma_f32_16x16x32_bf16(false, (A), false, (B), (short)0, (C), false, false)

// ---------------------------------------------------------------------------
// A-fragment loader (f32 memory -> bf16), contiguous k order, matching stage_w's B layout:
//   lane L holds row M=L%16;
//   lanes 0-15 : halves 0-15 = K kb+0..15
//   lanes 16-31: halves 0-15 = K kb+16..31
// k0 is a multiple of 8 floats -> 32B-aligned float4 loads (global_load_b128).
// ---------------------------------------------------------------------------
__device__ __forceinline__ v16bf load_a_frag(const float* __restrict__ row,
                                             int kb, int hi, float scale) {
  const int k0 = kb + (hi ? 16 : 0);
  const float4 p0 = *(const float4*)(row + k0);
  const float4 p1 = *(const float4*)(row + k0 + 4);
  const float4 p2 = *(const float4*)(row + k0 + 8);
  const float4 p3 = *(const float4*)(row + k0 + 12);
  v16bf a;
  a[0]  = (__bf16)(p0.x * scale); a[1]  = (__bf16)(p0.y * scale);
  a[2]  = (__bf16)(p0.z * scale); a[3]  = (__bf16)(p0.w * scale);
  a[4]  = (__bf16)(p1.x * scale); a[5]  = (__bf16)(p1.y * scale);
  a[6]  = (__bf16)(p1.z * scale); a[7]  = (__bf16)(p1.w * scale);
  a[8]  = (__bf16)(p2.x * scale); a[9]  = (__bf16)(p2.y * scale);
  a[10] = (__bf16)(p2.z * scale); a[11] = (__bf16)(p2.w * scale);
  a[12] = (__bf16)(p3.x * scale); a[13] = (__bf16)(p3.y * scale);
  a[14] = (__bf16)(p3.z * scale); a[15] = (__bf16)(p3.w * scale);
  return a;
}

// Stage a KxN (N=64) f32 weight matrix into LDS as bf16, pre-swizzled into
// per-lane WMMA B-fragment order (B 32x16: lane L holds col N=L%16, lanes 0-15
// K kb+0..15, lanes 16-31 K kb+16..31). Fragment f = kt*4+nt; lane slot holds
// 16 contiguous halves -> inner loop reads one 32B v16bf (2x ds_load_b128).
__device__ __forceinline__ void stage_w(const float* __restrict__ W, int K,
                                        __bf16* __restrict__ dst, int tid) {
  for (int pos = tid; pos < K * H; pos += 256) {
    const int k  = pos >> 6;            // row (K dim), coalesced reads over cols
    const int c  = pos & 63;            // col (N dim)
    const int kt = k >> 5;
    const int kk = k & 31;
    const int hi = kk >> 4;
    const int j  = kk & 15;
    const int nt = c >> 4;
    const int ln = hi * 16 + (c & 15);
    dst[(((kt << 2) + nt) * 32 + ln) * 16 + j] = (__bf16)W[pos];
  }
}

// ---------------------------------------------------------------------------
// Fused GEMM: Y[M x 64] = epi( A1[M x K1]*diag(1/max(cnt,1))? @ W1 + bias
//                              (+ A2[M x 64] @ W2) )
// One wave -> 16 rows x 64 cols (4 accumulators of v_wmma_f32_16x16x32_bf16).
// Weights staged once per block into LDS in fragment order; the four B
// fragments per k-step live in distinct temporaries so the ds_load_b128s can
// issue as one clause and overlap with earlier WMMAs (no per-op dscnt 0 stall).
// EPI: 0 = relu, 1 = row-wise L2 normalize (eps 1e-12).
// ---------------------------------------------------------------------------
template <int EPI, bool HAS2, bool SCALE>
__global__ __launch_bounds__(256, 2)
void gemm_h64(const float* __restrict__ A1, const float* __restrict__ cnt,
              const float* __restrict__ W1, const float* __restrict__ bias,
              const float* __restrict__ A2, const float* __restrict__ W2,
              float* __restrict__ Y, int M, int K1) {
  // max: K1=128 no-W2 -> 8192 halves; K1=64 + W2(64) -> 8192 halves (16 KB)
  __shared__ __align__(32) __bf16 wsm[8192];

  stage_w(W1, K1, wsm, threadIdx.x);
  const int w2off = (K1 >> 5) * 2048;               // halves used by W1
  if (HAS2) stage_w(W2, H, wsm + w2off, threadIdx.x);
  __syncthreads();

  const int lane    = threadIdx.x & 31;
  const int wv      = threadIdx.x >> 5;
  const int rowBase = (blockIdx.x * 8 + wv) * 16;
  if (rowBase >= M) return;            // wave-uniform; EXEC all-ones for WMMA
  const int n  = lane & 15;
  const int hi = lane >> 4;
  const int mA = rowBase + n;          // A-layout: lane holds row M = lane%16

  float scale = 1.0f;
  if (SCALE) scale = 1.0f / fmaxf(cnt[mA], 1.0f);

  const v16bf* __restrict__ bf1 = (const v16bf*)wsm;            // [frag][lane]
  const v16bf* __restrict__ bf2 = (const v16bf*)(wsm + w2off);

  v8f acc0 = {}, acc1 = {}, acc2 = {}, acc3 = {};

  const float* rowA1 = A1 + (size_t)mA * (size_t)K1;
  for (int kb = 0; kb < K1; kb += 32) {
    const v16bf a = load_a_frag(rowA1, kb, hi, scale);
    const int f = (kb >> 5) << 2;
    const v16bf w0 = bf1[(f + 0) * 32 + lane];
    const v16bf w1 = bf1[(f + 1) * 32 + lane];
    const v16bf w2 = bf1[(f + 2) * 32 + lane];
    const v16bf w3 = bf1[(f + 3) * 32 + lane];
    acc0 = WMMA_BF16(a, w0, acc0);
    acc1 = WMMA_BF16(a, w1, acc1);
    acc2 = WMMA_BF16(a, w2, acc2);
    acc3 = WMMA_BF16(a, w3, acc3);
  }
  if (HAS2) {
    const float* rowA2 = A2 + (size_t)mA * H;
    for (int kb = 0; kb < H; kb += 32) {
      const v16bf a = load_a_frag(rowA2, kb, hi, 1.0f);
      const int f = (kb >> 5) << 2;
      const v16bf w0 = bf2[(f + 0) * 32 + lane];
      const v16bf w1 = bf2[(f + 1) * 32 + lane];
      const v16bf w2 = bf2[(f + 2) * 32 + lane];
      const v16bf w3 = bf2[(f + 3) * 32 + lane];
      acc0 = WMMA_BF16(a, w0, acc0);
      acc1 = WMMA_BF16(a, w1, acc1);
      acc2 = WMMA_BF16(a, w2, acc2);
      acc3 = WMMA_BF16(a, w3, acc3);
    }
  }

  const float b0 = bias[n], b1 = bias[n + 16], b2 = bias[n + 32], b3 = bias[n + 48];

  // C/D layout: VGPR r holds row r (lanes 0-15) / row r+8 (lanes 16-31), col = lane%16.
#pragma unroll
  for (int r = 0; r < 8; ++r) {
    float v0 = acc0[r] + b0;
    float v1 = acc1[r] + b1;
    float v2 = acc2[r] + b2;
    float v3 = acc3[r] + b3;
    if (EPI == 0) {
      v0 = fmaxf(v0, 0.f); v1 = fmaxf(v1, 0.f);
      v2 = fmaxf(v2, 0.f); v3 = fmaxf(v3, 0.f);
    } else {
      float ss = v0 * v0 + v1 * v1 + v2 * v2 + v3 * v3;
      ss += __shfl_xor(ss, 1);  ss += __shfl_xor(ss, 2);
      ss += __shfl_xor(ss, 4);  ss += __shfl_xor(ss, 8);   // stays within 16-lane half
      const float inv = 1.0f / fmaxf(sqrtf(ss), 1e-12f);
      v0 *= inv; v1 *= inv; v2 *= inv; v3 *= inv;
    }
    float* yrow = Y + (size_t)(rowBase + r + hi * 8) * H + n;
    yrow[0] = v0; yrow[16] = v1; yrow[32] = v2; yrow[48] = v3;
  }
}

// ---------------------------------------------------------------------------
// Scatter-sum for mean aggregation: 16 threads per edge, float4 per thread.
// sum[dst] += X[src]; cnt[dst] += 1. (f32 global atomics -> L2 atomic units)
// ---------------------------------------------------------------------------
__global__ __launch_bounds__(256)
void scatter_accum(const float* __restrict__ X, const int* __restrict__ src,
                   const int* __restrict__ dst, float* __restrict__ sum,
                   float* __restrict__ cnt, int E) {
  const long long t = (long long)blockIdx.x * blockDim.x + threadIdx.x;
  const int e = (int)(t >> 4);
  const int p = (int)(t & 15);
  if (e >= E) return;
  const int s = src[e], d = dst[e];
  const float4 v = *(const float4*)(X + (size_t)s * H + p * 4);
  float* o = sum + (size_t)d * H + p * 4;
  atomicAdd(o + 0, v.x);
  atomicAdd(o + 1, v.y);
  atomicAdd(o + 2, v.z);
  atomicAdd(o + 3, v.w);
  if (p == 0) atomicAdd(cnt + d, 1.0f);
}

// ---------------------------------------------------------------------------
// HeteroConv mean (optional) -> relu -> LayerNorm. One wave per row (H=64).
// ---------------------------------------------------------------------------
__global__ __launch_bounds__(256)
void post_ln(const float* __restrict__ o1, const float* __restrict__ o2, int has2,
             const float* __restrict__ g, const float* __restrict__ bt,
             float* __restrict__ out, int M) {
  const int row  = (int)(((long long)blockIdx.x * blockDim.x + threadIdx.x) >> 5);
  const int lane = threadIdx.x & 31;
  if (row >= M) return;
  const size_t base = (size_t)row * H;
  float v0 = o1[base + lane];
  float v1 = o1[base + lane + 32];
  if (has2) {
    v0 = (v0 + o2[base + lane])      * 0.5f;
    v1 = (v1 + o2[base + lane + 32]) * 0.5f;
  }
  v0 = fmaxf(v0, 0.f);
  v1 = fmaxf(v1, 0.f);
  float s1 = v0 + v1;
  float s2 = v0 * v0 + v1 * v1;
#pragma unroll
  for (int m = 1; m < 32; m <<= 1) {
    s1 += __shfl_xor(s1, m);
    s2 += __shfl_xor(s2, m);
  }
  const float mean = s1 * (1.0f / H);
  const float var  = s2 * (1.0f / H) - mean * mean;
  const float inv  = rsqrtf(var + 1e-5f);
  out[base + lane]      = (v0 - mean) * inv * g[lane]      + bt[lane];
  out[base + lane + 32] = (v1 - mean) * inv * g[lane + 32] + bt[lane + 32];
}

// Final regression head: out[i] = dot(hs[i], Wreg) + breg. One wave per row.
__global__ __launch_bounds__(256)
void rowdot64(const float* __restrict__ X, const float* __restrict__ w,
              const float* __restrict__ b, float* __restrict__ out, int M) {
  const int row  = (int)(((long long)blockIdx.x * blockDim.x + threadIdx.x) >> 5);
  const int lane = threadIdx.x & 31;
  if (row >= M) return;
  const float* x = X + (size_t)row * H;
  float s = x[lane] * w[lane] + x[lane + 32] * w[lane + 32];
#pragma unroll
  for (int m = 1; m < 32; m <<= 1) s += __shfl_xor(s, m);
  if (lane == 0) out[row] = s + b[0];
}

__global__ __launch_bounds__(256)
void zero_f32(float* __restrict__ p, size_t n) {
  size_t i = (size_t)blockIdx.x * blockDim.x + threadIdx.x;
  const size_t stride = (size_t)gridDim.x * blockDim.x;
  for (; i < n; i += stride) p[i] = 0.0f;
}

// ---------------------------------------------------------------------------
extern "C" void kernel_launch(void* const* d_in, const int* in_sizes, int n_in,
                              void* d_out, int out_size, void* d_ws, size_t ws_size,
                              hipStream_t stream) {
  const float* x_stay  = (const float*)d_in[0];
  const float* x_pat   = (const float*)d_in[1];
  const float* Wp_stay = (const float*)d_in[2];
  const float* bp_stay = (const float*)d_in[3];
  const float* Wp_pat  = (const float*)d_in[4];
  const float* bp_pat  = (const float*)d_in[5];
  const float* Wl      = (const float*)d_in[6];   // [2][3][64][64]
  const float* bl      = (const float*)d_in[7];   // [2][3][64]
  const float* Wr      = (const float*)d_in[8];   // [2][3][64][64]
  const float* ln_g    = (const float*)d_in[9];   // [2][64]
  const float* ln_b    = (const float*)d_in[10];  // [2][64]
  const float* Wreg    = (const float*)d_in[11];  // [64]
  const float* breg    = (const float*)d_in[12];  // [1]
  const int* e1_src = (const int*)d_in[13];
  const int* e1_dst = (const int*)d_in[14];
  const int* e2_src = (const int*)d_in[15];
  const int* e2_dst = (const int*)d_in[16];
  const int* e3_src = (const int*)d_in[17];
  const int* e3_dst = (const int*)d_in[18];

  // Workspace layout (floats)
  float* ws   = (float*)d_ws;
  float* hs   = ws;                             // N_STAY*64
  float* hp   = hs   + (size_t)N_STAY * H;      // N_PAT*64
  float* aggs = hp   + (size_t)N_PAT  * H;      // N_STAY*64 (shared agg buffer)
  float* cnt  = aggs + (size_t)N_STAY * H;      // N_STAY
  float* o1   = cnt  + (size_t)N_STAY;          // N_PAT*64
  float* o2   = o1   + (size_t)N_PAT  * H;      // N_STAY*64
  float* o3   = o2   + (size_t)N_STAY * H;      // N_STAY*64

  const int gS = (N_STAY / 16 + 7) / 8;  // GEMM blocks, stay rows
  const int gP = (N_PAT  / 16 + 7) / 8;  // GEMM blocks, patient rows
  const int gE = (NEDGE * 16 + 255) / 256;
  const int wS = N_STAY * 32 / 256;      // wave-per-row blocks, stays
  const int wP = N_PAT  * 32 / 256;

  // Input projections + relu
  gemm_h64<0, false, false><<<gS, 256, 0, stream>>>(
      x_stay, nullptr, Wp_stay, bp_stay, nullptr, nullptr, hs, N_STAY, 128);
  gemm_h64<0, false, false><<<gP, 256, 0, stream>>>(
      x_pat, nullptr, Wp_pat, bp_pat, nullptr, nullptr, hp, N_PAT, 64);

  for (int l = 0; l < 2; ++l) {
    const float* Wl0 = Wl + (size_t)(l * 3 + 0) * H * H;
    const float* Wl1 = Wl + (size_t)(l * 3 + 1) * H * H;
    const float* Wl2 = Wl + (size_t)(l * 3 + 2) * H * H;
    const float* Wr0 = Wr + (size_t)(l * 3 + 0) * H * H;
    const float* Wr1 = Wr + (size_t)(l * 3 + 1) * H * H;
    const float* Wr2 = Wr + (size_t)(l * 3 + 2) * H * H;
    const float* bl0 = bl + (size_t)(l * 3 + 0) * H;
    const float* bl1 = bl + (size_t)(l * 3 + 1) * H;
    const float* bl2 = bl + (size_t)(l * 3 + 2) * H;

    // conv1: stay -> patient
    zero_f32<<<2048, 256, 0, stream>>>(aggs, (size_t)N_PAT * H);
    zero_f32<<<256,  256, 0, stream>>>(cnt,  (size_t)N_PAT);
    scatter_accum<<<gE, 256, 0, stream>>>(hs, e1_src, e1_dst, aggs, cnt, NEDGE);
    gemm_h64<1, true, true><<<gP, 256, 0, stream>>>(
        aggs, cnt, Wl0, bl0, hp, Wr0, o1, N_PAT, 64);

    // conv2: patient -> stay
    zero_f32<<<2048, 256, 0, stream>>>(aggs, (size_t)N_STAY * H);
    zero_f32<<<256,  256, 0, stream>>>(cnt,  (size_t)N_STAY);
    scatter_accum<<<gE, 256, 0, stream>>>(hp, e2_src, e2_dst, aggs, cnt, NEDGE);
    gemm_h64<1, true, true><<<gS, 256, 0, stream>>>(
        aggs, cnt, Wl1, bl1, hs, Wr1, o2, N_STAY, 64);

    // conv3: stay -> stay
    zero_f32<<<2048, 256, 0, stream>>>(aggs, (size_t)N_STAY * H);
    zero_f32<<<256,  256, 0, stream>>>(cnt,  (size_t)N_STAY);
    scatter_accum<<<gE, 256, 0, stream>>>(hs, e3_src, e3_dst, aggs, cnt, NEDGE);
    gemm_h64<1, true, true><<<gS, 256, 0, stream>>>(
        aggs, cnt, Wl2, bl2, hs, Wr2, o3, N_STAY, 64);

    // HeteroConv mean + relu + LayerNorm (hs/hp inputs fully consumed above)
    post_ln<<<wS, 256, 0, stream>>>(o2, o3, 1, ln_g + l * H, ln_b + l * H, hs, N_STAY);
    post_ln<<<wP, 256, 0, stream>>>(o1, o1, 0, ln_g + l * H, ln_b + l * H, hp, N_PAT);
  }

  rowdot64<<<wS, 256, 0, stream>>>(hs, Wreg, breg, (float*)d_out, N_STAY);
}
